// Attention_74199855005770
// MI455X (gfx1250) — matrix operand order repaired and checked by reading, not compiled, not run on hardware
//
#include <hip/hip_runtime.h>
#include <hip/hip_bf16.h>
#include <cstdint>

#define B_   64
#define T_   2048
#define ENC_ 512
#define KQ_  256
#define VD_  256
#define MT_  4          // 16-row M tiles per block -> M = 64
#define NEG_SLOPE_ 0.2f

typedef __attribute__((ext_vector_type(16))) __bf16 v16bf;
typedef __attribute__((ext_vector_type(8)))  __bf16 v8bf;
typedef __attribute__((ext_vector_type(8)))  float  v8f;

__device__ __forceinline__ float leaky(float x) {
    return x >= 0.0f ? x : NEG_SLOPE_ * x;
}

// ---------------------------------------------------------------------------
// Pack a [512,256] f32 row-major weight matrix into the per-lane bf16 register
// image of the WMMA B operand (32x16 K-slab per tile).
// B order used here (contiguous K, as in A): lane = N (mod 16); lanes 0-15 hold K=0..15 of the
// 32-K slab, lanes 16-31 hold K=16..31; half h within a lane maps K in order.
// out index = ((s*16 + n)*32 + lane)*16 + h
// ---------------------------------------------------------------------------
__global__ __launch_bounds__(256) void pack_weights_kernel(
    const float* __restrict__ W, __bf16* __restrict__ out)
{
    int idx = blockIdx.x * 256 + threadIdx.x;       // 0 .. 131071
    int h = idx & 15;
    int l = (idx >> 4) & 31;
    int n = (idx >> 9) & 15;
    int s = idx >> 13;
    int N = n * 16 + (l & 15);
    int K = s * 32 + ((l < 16) ? h : (16 + h));
    out[idx] = (__bf16)W[K * KQ_ + N];
}

// ---------------------------------------------------------------------------
// query = leaky(decoder_feat @ Wq + bq)   [64, 256]  (tiny)
// ---------------------------------------------------------------------------
__global__ __launch_bounds__(256) void query_kernel(
    const float* __restrict__ dec, const float* __restrict__ Wq,
    const float* __restrict__ bq, float* __restrict__ q)
{
    int b = blockIdx.x;
    int k = threadIdx.x;
    __shared__ float drow[ENC_];
    for (int i = threadIdx.x; i < ENC_; i += 256) drow[i] = dec[b * ENC_ + i];
    __syncthreads();
    float acc = bq[k];
    for (int e = 0; e < ENC_; ++e) acc += drow[e] * Wq[e * KQ_ + k];
    q[b * KQ_ + k] = leaky(acc);
}

// ---------------------------------------------------------------------------
// Main WMMA kernel. Block = 256 threads (8 waves), M = 64 rows of T.
//   - enc K-slab (64x32 f32) cooperatively loaded, converted to bf16, staged
//     in LDS in the A-fragment register layout (double buffered).
//   - wave w owns columns [32w, 32w+32) of both K and V projections:
//     16 accumulator tiles, 16 wmma per k-step, each B fragment reused 4x.
//   - epilogue: bias + leaky, energy = q.K reduced via shfl + LDS,
//     V written to workspace as bf16.
// A-fragment order used here (16x32, contiguous K): lanes 0-15: row M=lane, halves 0-7 ->
// K+0..7, halves 8-15 -> K+8..15; lanes 16-31: row M=lane-16, halves 0-7 ->
// K+16..23, halves 8-15 -> K+24..31.
// ---------------------------------------------------------------------------
__global__ __launch_bounds__(256) void kv_energy_kernel(
    const float*  __restrict__ enc,
    const __bf16* __restrict__ WkP,
    const __bf16* __restrict__ WvP,
    const float*  __restrict__ bk,
    const float*  __restrict__ bv,
    const float*  __restrict__ q,
    float*        __restrict__ energy,
    __bf16*       __restrict__ value)
{
    int b  = blockIdx.x >> 5;               // 32 blocks per batch
    int t0 = (blockIdx.x & 31) << 6;        // 64 rows per block
    int lane = threadIdx.x & 31;
    int wave = threadIdx.x >> 5;            // 0..7

    __shared__ v16bf atile[2 * MT_ * 32];   // 2 x 4KB, 32B aligned fragments
    __shared__ float esh[MT_ * 16];

    // -------- cooperative staging setup (one 8-half group per thread) -----
    int g     = threadIdx.x;
    int smt   = g >> 6;                     // M tile 0..3
    int sl    = (g >> 1) & 31;              // lane slot 0..31
    int shalf = g & 1;                      // 0: halves 0-7, 1: halves 8-15
    int srow  = t0 + smt * 16 + (sl & 15);
    int skofs = ((sl < 16) ? 0 : 16) + shalf * 8;
    const float* sgp = enc + ((size_t)(b * T_ + srow)) * ENC_ + skofs;
    __bf16* sls = (__bf16*)atile + ((size_t)(smt * 32 + sl)) * 16 + shalf * 8;

    v8f accK[MT_][2], accV[MT_][2];
    v8f vzero = {};
#pragma unroll
    for (int mt = 0; mt < MT_; ++mt)
#pragma unroll
        for (int j = 0; j < 2; ++j) { accK[mt][j] = vzero; accV[mt][j] = vzero; }

    int nt0 = wave * 2;                     // this wave's first N-tile

    // stage slab 0
    {
        float4 x0 = *(const float4*)(sgp);
        float4 x1 = *(const float4*)(sgp + 4);
        v8bf pk;
        pk[0] = (__bf16)x0.x; pk[1] = (__bf16)x0.y;
        pk[2] = (__bf16)x0.z; pk[3] = (__bf16)x0.w;
        pk[4] = (__bf16)x1.x; pk[5] = (__bf16)x1.y;
        pk[6] = (__bf16)x1.z; pk[7] = (__bf16)x1.w;
        *(v8bf*)sls = pk;
    }
    __syncthreads();

    for (int s = 0; s < 16; ++s) {          // K = 512 -> 16 slabs of 32
        // stage next slab into the other buffer (overlaps with wmma below)
        if (s + 1 < 16) {
            float4 x0 = *(const float4*)(sgp + (s + 1) * 32);
            float4 x1 = *(const float4*)(sgp + (s + 1) * 32 + 4);
            v8bf pk;
            pk[0] = (__bf16)x0.x; pk[1] = (__bf16)x0.y;
            pk[2] = (__bf16)x0.z; pk[3] = (__bf16)x0.w;
            pk[4] = (__bf16)x1.x; pk[5] = (__bf16)x1.y;
            pk[6] = (__bf16)x1.z; pk[7] = (__bf16)x1.w;
            *(v8bf*)(sls + ((s + 1) & 1) * (MT_ * 32 * 16)) = pk;
        }

        // B fragments for this wave's 4 output tiles (L2-resident weights)
        size_t o0 = ((size_t)((s * 16 + nt0)     * 32 + lane)) << 4;
        size_t o1 = ((size_t)((s * 16 + nt0 + 1) * 32 + lane)) << 4;
        v16bf bK0 = *(const v16bf*)(WkP + o0);
        v16bf bK1 = *(const v16bf*)(WkP + o1);
        v16bf bV0 = *(const v16bf*)(WvP + o0);
        v16bf bV1 = *(const v16bf*)(WvP + o1);

        const v16bf* abuf = atile + (s & 1) * (MT_ * 32);
#pragma unroll
        for (int mt = 0; mt < MT_; ++mt) {
            v16bf a = abuf[mt * 32 + lane];
            accK[mt][0] = __builtin_amdgcn_wmma_f32_16x16x32_bf16(
                false, a, false, bK0, (short)0, accK[mt][0], false, false);
            accK[mt][1] = __builtin_amdgcn_wmma_f32_16x16x32_bf16(
                false, a, false, bK1, (short)0, accK[mt][1], false, false);
            accV[mt][0] = __builtin_amdgcn_wmma_f32_16x16x32_bf16(
                false, a, false, bV0, (short)0, accV[mt][0], false, false);
            accV[mt][1] = __builtin_amdgcn_wmma_f32_16x16x32_bf16(
                false, a, false, bV1, (short)0, accV[mt][1], false, false);
        }
        __syncthreads();
    }

    // -------- epilogue --------
    // C layout: VGPR r, lanes 0-15 -> (row r, col = nt*16+lane),
    //           lanes 16-31 -> (row r+8, col = nt*16+lane-16).
    if (threadIdx.x < MT_ * 16) esh[threadIdx.x] = 0.0f;
    __syncthreads();

    int trow_base = (lane < 16) ? 0 : 8;
#pragma unroll
    for (int mt = 0; mt < MT_; ++mt) {
        float ep[8];
#pragma unroll
        for (int r = 0; r < 8; ++r) ep[r] = 0.0f;
#pragma unroll
        for (int j = 0; j < 2; ++j) {
            int col = (nt0 + j) * 16 + (lane & 15);
            float qv  = q[b * KQ_ + col];
            float bkv = bk[col];
            float bvv = bv[col];
#pragma unroll
            for (int r = 0; r < 8; ++r) {
                float kv = leaky(accK[mt][j][r] + bkv);
                ep[r] += kv * qv;
                float vv = leaky(accV[mt][j][r] + bvv);
                int t = t0 + mt * 16 + trow_base + r;
                value[((size_t)(b * T_) + t) * VD_ + col] = (__bf16)vv;
            }
        }
#pragma unroll
        for (int r = 0; r < 8; ++r) {
            float v = ep[r];
            v += __shfl_xor(v, 1, 32);
            v += __shfl_xor(v, 2, 32);
            v += __shfl_xor(v, 4, 32);
            v += __shfl_xor(v, 8, 32);
            if (lane == 0)  atomicAdd(&esh[mt * 16 + r], v);
            if (lane == 16) atomicAdd(&esh[mt * 16 + 8 + r], v);
        }
    }
    __syncthreads();
    if (threadIdx.x < MT_ * 16)
        energy[(size_t)b * T_ + t0 + threadIdx.x] = esh[threadIdx.x];
}

// ---------------------------------------------------------------------------
// Masked softmax: score = exp(e-max)*mask / sum(exp*mask)  (the full-softmax
// normalization cancels against the renormalization in the reference).
// ---------------------------------------------------------------------------
__global__ __launch_bounds__(256) void softmax_kernel(
    const float* __restrict__ energy,
    const int*   __restrict__ seq_len,
    float*       __restrict__ score_out)
{
    int b   = blockIdx.x;
    int tid = threadIdx.x;
    int L   = seq_len[b];                    // 1 <= L <= T
    __shared__ float ex[T_];
    __shared__ float red[256];

    const float* eb = energy + (size_t)b * T_;

    float m = -3.4e38f;
    for (int t = tid; t < T_; t += 256)
        if (t < L) m = fmaxf(m, eb[t]);
    red[tid] = m;
    __syncthreads();
    for (int s2 = 128; s2 > 0; s2 >>= 1) {
        if (tid < s2) red[tid] = fmaxf(red[tid], red[tid + s2]);
        __syncthreads();
    }
    float gmax = red[0];
    __syncthreads();

    float ssum = 0.0f;
    for (int t = tid; t < T_; t += 256) {
        float e = (t < L) ? __expf(eb[t] - gmax) : 0.0f;
        ex[t] = e;
        ssum += e;
    }
    red[tid] = ssum;
    __syncthreads();
    for (int s2 = 128; s2 > 0; s2 >>= 1) {
        if (tid < s2) red[tid] += red[tid + s2];
        __syncthreads();
    }
    float inv = 1.0f / red[0];

    for (int t = tid; t < T_; t += 256)
        score_out[(size_t)b * T_ + t] = ex[t] * inv;
}

// ---------------------------------------------------------------------------
// context partials: ctx[b,v] = sum_t score[b,t] * V[b,t,v], split over 8
// T-chunks per batch; deterministic two-stage reduction (no float atomics).
// ---------------------------------------------------------------------------
__global__ __launch_bounds__(256) void context_partial_kernel(
    const float*  __restrict__ score,
    const __bf16* __restrict__ value,
    const int*    __restrict__ seq_len,
    float*        __restrict__ partial)
{
    int b   = blockIdx.x >> 3;
    int c   = blockIdx.x & 7;               // chunk of 256 t-steps
    int tid = threadIdx.x;
    int L   = seq_len[b];
    int tstart = c * 256;
    float acc = 0.0f;
    if (tstart < L) {
        int tend = (tstart + 256 < L) ? (tstart + 256) : L;
        const float*  sb = score + (size_t)b * T_;
        const __bf16* vb = value + (size_t)b * T_ * VD_;
        float a0 = 0.f, a1 = 0.f, a2 = 0.f, a3 = 0.f;
        int t = tstart;
        for (; t + 3 < tend; t += 4) {
            a0 += sb[t + 0] * (float)vb[(size_t)(t + 0) * VD_ + tid];
            a1 += sb[t + 1] * (float)vb[(size_t)(t + 1) * VD_ + tid];
            a2 += sb[t + 2] * (float)vb[(size_t)(t + 2) * VD_ + tid];
            a3 += sb[t + 3] * (float)vb[(size_t)(t + 3) * VD_ + tid];
        }
        for (; t < tend; ++t)
            a0 += sb[t] * (float)vb[(size_t)t * VD_ + tid];
        acc = (a0 + a1) + (a2 + a3);
    }
    partial[(size_t)blockIdx.x * VD_ + tid] = acc;   // zero for skipped chunks
}

__global__ __launch_bounds__(256) void context_reduce_kernel(
    const float* __restrict__ partial, float* __restrict__ ctx)
{
    int b = blockIdx.x;
    int tid = threadIdx.x;
    float s = 0.0f;
#pragma unroll
    for (int c = 0; c < 8; ++c)
        s += partial[((size_t)(b * 8 + c)) * VD_ + tid];
    ctx[b * VD_ + tid] = s;
}

// ---------------------------------------------------------------------------
extern "C" void kernel_launch(void* const* d_in, const int* in_sizes, int n_in,
                              void* d_out, int out_size, void* d_ws, size_t ws_size,
                              hipStream_t stream) {
    const float* dec = (const float*)d_in[0];
    const float* enc = (const float*)d_in[1];
    const float* Wq  = (const float*)d_in[2];
    const float* bq  = (const float*)d_in[3];
    const float* Wk  = (const float*)d_in[4];
    const float* bk  = (const float*)d_in[5];
    const float* Wv  = (const float*)d_in[6];
    const float* bv  = (const float*)d_in[7];
    const int*   seq = (const int*)d_in[8];

    char* ws = (char*)d_ws;
    __bf16* WkP     = (__bf16*)(ws);                          // 256 KB
    __bf16* WvP     = (__bf16*)(ws + (size_t)262144);         // 256 KB
    float*  qbuf    = (float*) (ws + (size_t)524288);         // 64 KB
    float*  energy  = (float*) (ws + (size_t)589824);         // 512 KB
    __bf16* value   = (__bf16*)(ws + (size_t)1114112);        // 64 MB
    float*  partial = (float*) (ws + (size_t)68222976);       // 512 KB

    float* score_out = (float*)d_out;                         // [64, 2048]
    float* ctx_out   = score_out + (size_t)B_ * T_;           // [64, 256]

    pack_weights_kernel<<<dim3(512), dim3(256), 0, stream>>>(Wk, WkP);
    pack_weights_kernel<<<dim3(512), dim3(256), 0, stream>>>(Wv, WvP);
    query_kernel<<<dim3(B_), dim3(256), 0, stream>>>(dec, Wq, bq, qbuf);
    kv_energy_kernel<<<dim3(B_ * (T_ / (16 * MT_))), dim3(256), 0, stream>>>(
        enc, WkP, WvP, bk, bv, qbuf, energy, value);
    softmax_kernel<<<dim3(B_), dim3(256), 0, stream>>>(energy, seq, score_out);
    context_partial_kernel<<<dim3(B_ * 8), dim3(256), 0, stream>>>(
        score_out, value, seq, partial);
    context_reduce_kernel<<<dim3(B_), dim3(256), 0, stream>>>(partial, ctx_out);
}
